// multi_heads_self_attention_2327872274493
// MI455X (gfx1250) — matrix operand order repaired and checked by reading, not compiled, not run on hardware
//
// MI455X (gfx1250, CDNA5) multi-head self-attention + layernorm.
// bf16 v_wmma_f32_16x16x32_bf16 for all GEMMs (8x K-depth per issue vs f32 WMMA),
// f32 accumulate, f32 softmax/layernorm. Attention is 3-pass because the full
// attn matrix is a required output (537 MB -> ~25us/pass HBM floor @23.3TB/s).
// This revision: double-buffered LDS (1 barrier per K-step instead of 2) and
// CDNA5 GLOBAL_LOAD_ASYNC_TO_LDS_B128 staging (ASYNCcnt) for bf16 operands.
#include <hip/hip_runtime.h>
#include <hip/hip_bf16.h>
#include <stdint.h>

typedef __attribute__((ext_vector_type(16))) __bf16 v16bf;
typedef __attribute__((ext_vector_type(8)))  float  v8f;

static constexpr int Bsz = 8, Ssz = 1024, Fsz = 1024, Hsz = 16, Dsz = 64;
static constexpr float SCALE = 0.5f;   // (D // H) ** -0.5 = 4 ** -0.5

// ---------------- f32 -> bf16 conversion (4 elems / thread) ----------------
__global__ __launch_bounds__(256) void cvt_f32_bf16(const float* __restrict__ src,
                                                    __bf16* __restrict__ dst, int n4) {
  int i = blockIdx.x * blockDim.x + threadIdx.x;
  if (i >= n4) return;
  float4 f = ((const float4*)src)[i];
  union { __bf16 h[4]; unsigned long long u; } r;
  r.h[0] = (__bf16)f.x; r.h[1] = (__bf16)f.y;
  r.h[2] = (__bf16)f.z; r.h[3] = (__bf16)f.w;
  ((unsigned long long*)dst)[i] = r.u;
}

// lds byte address = low 32 bits of generic pointer (ISA 10.2 LDS aperture)
__device__ __forceinline__ unsigned lds_addr(const void* p) {
  return (unsigned)(uintptr_t)p;
}
// CDNA5 async copy: 16 bytes memory -> LDS, tracked by ASYNCcnt.
// INST_OFFSET is added to BOTH lds and global address (ISA 10.7 pseudocode).
__device__ __forceinline__ void async_cp16(unsigned lds, unsigned long long gaddr) {
  asm volatile("global_load_async_to_lds_b128 %0, %1, off"
               :: "v"(lds), "v"(gaddr) : "memory");
}
__device__ __forceinline__ void async_cp16_off16(unsigned lds, unsigned long long gaddr) {
  asm volatile("global_load_async_to_lds_b128 %0, %1, off offset:16"
               :: "v"(lds), "v"(gaddr) : "memory");
}
__device__ __forceinline__ void wait_async0() {
  asm volatile("s_wait_asynccnt 0x0" ::: "memory");
}

// ---------------- NT GEMM: C[m,n] = alpha * sum_k A[m,k]*B[n,k] (+bias[n]) ----
// Both operands row-major over K. Block tile BM x BN, K-step 32, LDS double-buffered.
// 256 threads = 8 waves arranged WROWS x WCOLS; per-wave tile (BM/WROWS)x(BN/WCOLS).
// Batched: operand offset = (z/hdiv)*BS + (z%hdiv)*HS  (per-head addressing).
template<int BM, int BN, int WROWS, int WCOLS, bool A_F32, bool OUT_F32, bool BIAS>
__global__ __launch_bounds__(256)
void gemm_nt_wmma(const void* __restrict__ Abase, const __bf16* __restrict__ Bbase,
                  const float* __restrict__ bias, void* __restrict__ Cbase,
                  int lda, int ldb, int ldc, int K,
                  long long aBS, long long aHS, long long bBS, long long bHS,
                  long long cBS, long long cHS, int hdiv, float alpha) {
  constexpr int KT = 32;
  constexpr int LDSS = KT + 8;              // bf16 elems per LDS row (+16B pad)
  constexpr int WTM = BM / WROWS;
  constexpr int WTN = BN / WCOLS;
  constexpr int MT = WTM / 16;
  constexpr int NT = WTN / 16;

  __shared__ __bf16 sA[2][BM * LDSS];
  __shared__ __bf16 sB[2][BN * LDSS];

  const int tid    = threadIdx.x;
  const int wid    = tid >> 5;
  const int lane   = tid & 31;
  const int laneLo = lane & 15;
  const int laneHi = lane >> 4;
  const int waveR  = wid / WCOLS;
  const int waveC  = wid % WCOLS;

  const int z = blockIdx.z;
  const long long aOff = (long long)(z / hdiv) * aBS + (long long)(z % hdiv) * aHS;
  const long long bOff = (long long)(z / hdiv) * bBS + (long long)(z % hdiv) * bHS;
  const long long cOff = (long long)(z / hdiv) * cBS + (long long)(z % hdiv) * cHS;

  const int m0 = blockIdx.y * BM;
  const int n0 = blockIdx.x * BN;

  // global->LDS staging map: A tile is BM x 32 bf16 (16 elems/thread)
  const int arow  = tid >> 1;
  const int akoff = (tid & 1) * 16;
  const int brow  = (BN == 128) ? (tid >> 1) : (tid >> 2);
  const int bkoff = (BN == 128) ? (tid & 1) * 16 : (tid & 3) * 8;

  const __bf16* aSrcB = (const __bf16*)Abase + aOff + (long long)(m0 + arow) * lda + akoff;
  const float*  aSrcF = (const float*)Abase + aOff + (long long)(m0 + arow) * lda + akoff;
  const __bf16* bSrc  = Bbase + bOff + (long long)(n0 + brow) * ldb + bkoff;

  float4 aregF[4];

  // Issue stage: B (and bf16-A) go via GLOBAL_LOAD_ASYNC_TO_LDS_B128; f32-A
  // loads into regs (converted and stored in finishStage).
  auto issueStage = [&](int k0, int buf) {
    unsigned lB = lds_addr(&sB[buf][brow * LDSS + bkoff]);
    unsigned long long gB = (unsigned long long)(uintptr_t)(bSrc + k0);
    async_cp16(lB, gB);
    if (BN == 128) async_cp16_off16(lB, gB);
    if (A_F32) {
      const float* p = aSrcF + k0;
      aregF[0] = ((const float4*)p)[0]; aregF[1] = ((const float4*)p)[1];
      aregF[2] = ((const float4*)p)[2]; aregF[3] = ((const float4*)p)[3];
    } else {
      unsigned lA = lds_addr(&sA[buf][arow * LDSS + akoff]);
      unsigned long long gA = (unsigned long long)(uintptr_t)(aSrcB + k0);
      async_cp16(lA, gA);
      async_cp16_off16(lA, gA);
    }
  };
  auto finishStage = [&](int buf) {
    if (A_F32) {
      union { __bf16 h[16]; uint4 u[2]; } cv;
      const float* f = (const float*)aregF;
      #pragma unroll
      for (int i = 0; i < 16; ++i) cv.h[i] = (__bf16)f[i];
      *(uint4*)&sA[buf][arow * LDSS + akoff]     = cv.u[0];
      *(uint4*)&sA[buf][arow * LDSS + akoff + 8] = cv.u[1];
    }
    wait_async0();   // own async copies done -> safe to signal barrier
  };

  // A fragment (16-bit A 16x32): lane lo K0..15, hi K16..31 (matches B fragment)
  auto fragA = [&](int mt, int buf) -> v16bf {
    const __bf16* p = &sA[buf][(waveR * WTM + mt * 16 + laneLo) * LDSS];
    union { uint4 u[2]; v16bf v; } r;
    r.u[0] = *(const uint4*)(p + laneHi * 16);
    r.u[1] = *(const uint4*)(p + laneHi * 16 + 8);
    return r.v;
  };
  // B fragment (32x16, K x N): lane group lo holds K0..15, hi K16..31, N = laneLo
  auto fragB = [&](int nt, int buf) -> v16bf {
    const __bf16* p = &sB[buf][(waveC * WTN + nt * 16 + laneLo) * LDSS + laneHi * 16];
    union { uint4 u[2]; v16bf v; } r;
    r.u[0] = *(const uint4*)(p);
    r.u[1] = *(const uint4*)(p + 8);
    return r.v;
  };

  v8f acc[MT][NT];
  const v8f vz = {0.f, 0.f, 0.f, 0.f, 0.f, 0.f, 0.f, 0.f};
  #pragma unroll
  for (int mt = 0; mt < MT; ++mt)
    #pragma unroll
    for (int nt = 0; nt < NT; ++nt) acc[mt][nt] = vz;

  const int ktiles = K / KT;
  issueStage(0, 0);
  finishStage(0);
  __syncthreads();

  int cur = 0;
  for (int kt = 0; kt < ktiles; ++kt) {
    if (kt + 1 < ktiles) issueStage((kt + 1) * KT, cur ^ 1);
    if (kt + 2 < ktiles) {   // global_prefetch_b8 hint two tiles ahead
      if (A_F32) __builtin_prefetch((const void*)(aSrcF + (kt + 2) * KT), 0, 0);
      else       __builtin_prefetch((const void*)(aSrcB + (kt + 2) * KT), 0, 0);
      __builtin_prefetch((const void*)(bSrc + (kt + 2) * KT), 0, 0);
    }
    v16bf af[MT]; v16bf bf[NT];
    #pragma unroll
    for (int mt = 0; mt < MT; ++mt) af[mt] = fragA(mt, cur);
    #pragma unroll
    for (int nt = 0; nt < NT; ++nt) bf[nt] = fragB(nt, cur);
    #pragma unroll
    for (int mt = 0; mt < MT; ++mt)
      #pragma unroll
      for (int nt = 0; nt < NT; ++nt)
        acc[mt][nt] = __builtin_amdgcn_wmma_f32_16x16x32_bf16(
            false, af[mt], false, bf[nt], (short)0, acc[mt][nt], false, false);
    if (kt + 1 < ktiles) finishStage(cur ^ 1);
    __syncthreads();
    cur ^= 1;
  }

  // Epilogue (C layout: VGPR e -> M = e (+8 for hi lanes), N = laneLo)
  float*  Cf = (float*)Cbase + cOff;
  __bf16* Ch = (__bf16*)Cbase + cOff;
  #pragma unroll
  for (int mt = 0; mt < MT; ++mt)
    #pragma unroll
    for (int nt = 0; nt < NT; ++nt) {
      int col = n0 + waveC * WTN + nt * 16 + laneLo;
      float bv = BIAS ? bias[col] : 0.f;
      #pragma unroll
      for (int e = 0; e < 8; ++e) {
        int row = m0 + waveR * WTM + mt * 16 + laneHi * 8 + e;
        float v = acc[mt][nt][e] * alpha + bv;
        if (OUT_F32) Cf[(long long)row * ldc + col] = v;
        else         Ch[(long long)row * ldc + col] = (__bf16)v;
      }
    }
}

// ---------------- per-head transpose: vT[bh][d][s] = v[b][s][h*64+d] -----------
__global__ __launch_bounds__(256) void transpose_v(const __bf16* __restrict__ v,
                                                   __bf16* __restrict__ vT) {
  __shared__ __bf16 t[32][33];
  int z = blockIdx.z, b = z / Hsz, h = z % Hsz;
  int s0 = blockIdx.x * 32, d0 = blockIdx.y * 32;
  int tx = threadIdx.x & 31, ty = threadIdx.x >> 5;   // ty: 0..7
  #pragma unroll
  for (int i = 0; i < 4; ++i) {
    int s = s0 + ty + i * 8;
    t[ty + i * 8][tx] = v[(long long)b * Ssz * Fsz + (long long)s * Fsz + h * Dsz + d0 + tx];
  }
  __syncthreads();
  #pragma unroll
  for (int i = 0; i < 4; ++i) {
    int d = d0 + ty + i * 8;
    vT[(long long)z * Dsz * Ssz + (long long)d * Ssz + s0 + tx] = t[tx][ty + i * 8];
  }
}

// ---------------- wave-per-row softmax, in place, cols = 1024 ------------------
__global__ __launch_bounds__(256) void softmax_rows(float* __restrict__ attn, long long rows) {
  int wid = threadIdx.x >> 5, lane = threadIdx.x & 31;
  long long row = (long long)blockIdx.x * 8 + wid;
  if (row >= rows) return;
  float* p = attn + row * 1024ll;
  float v[32];
  float mx = -1e30f;
  #pragma unroll
  for (int j = 0; j < 32; ++j) { v[j] = p[lane + j * 32]; mx = fmaxf(mx, v[j]); }
  #pragma unroll
  for (int o = 16; o > 0; o >>= 1) mx = fmaxf(mx, __shfl_xor(mx, o, 32));
  float s = 0.f;
  #pragma unroll
  for (int j = 0; j < 32; ++j) { v[j] = __expf(v[j] - mx); s += v[j]; }
  #pragma unroll
  for (int o = 16; o > 0; o >>= 1) s += __shfl_xor(s, o, 32);
  float inv = 1.f / s;
  #pragma unroll
  for (int j = 0; j < 32; ++j) p[lane + j * 32] = v[j] * inv;
}

// ---------------- residual + layernorm, wave per row (F = 1024) ----------------
__global__ __launch_bounds__(256) void residual_layernorm(
    const float* __restrict__ resid, const float* __restrict__ y,
    const float* __restrict__ gamma, const float* __restrict__ beta,
    float* __restrict__ out, long long rows) {
  int wid = threadIdx.x >> 5, lane = threadIdx.x & 31;
  long long row = (long long)blockIdx.x * 8 + wid;
  if (row >= rows) return;
  const float* rp = resid + row * (long long)Fsz;
  const float* yp = y + row * (long long)Fsz;
  float x[32]; float s = 0.f;
  #pragma unroll
  for (int j = 0; j < 32; ++j) { int c = lane + j * 32; x[j] = rp[c] + yp[c]; s += x[j]; }
  #pragma unroll
  for (int o = 16; o > 0; o >>= 1) s += __shfl_xor(s, o, 32);
  float mu = s * (1.f / 1024.f);
  float var = 0.f;
  #pragma unroll
  for (int j = 0; j < 32; ++j) { float d = x[j] - mu; var += d * d; }
  #pragma unroll
  for (int o = 16; o > 0; o >>= 1) var += __shfl_xor(var, o, 32);
  float inv = rsqrtf(var * (1.f / 1024.f) + 1e-5f);
  float* op = out + row * (long long)Fsz;
  #pragma unroll
  for (int j = 0; j < 32; ++j) { int c = lane + j * 32; op[c] = (x[j] - mu) * inv * gamma[c] + beta[c]; }
}

// =============================== host launcher ================================
extern "C" void kernel_launch(void* const* d_in, const int* in_sizes, int n_in,
                              void* d_out, int out_size, void* d_ws, size_t ws_size,
                              hipStream_t stream) {
  const float* key   = (const float*)d_in[0];
  const float* value = (const float*)d_in[1];
  const float* query = (const float*)d_in[2];
  const float* Wq = (const float*)d_in[3];  const float* bq = (const float*)d_in[4];
  const float* Wk = (const float*)d_in[5];  const float* bk = (const float*)d_in[6];
  const float* Wv = (const float*)d_in[7];  const float* bv = (const float*)d_in[8];
  const float* Wo = (const float*)d_in[9];  const float* bo = (const float*)d_in[10];
  const float* gamma = (const float*)d_in[11];
  const float* beta  = (const float*)d_in[12];

  float* out  = (float*)d_out;                           // [B,S,F]
  float* attn = out + (long long)Bsz * Ssz * Fsz;        // [B*H,S,S]

  const size_t NX = (size_t)Bsz * Ssz * Fsz;             // 8M activations
  const size_t NW = (size_t)Fsz * Fsz;                   // 1M weights

  char* w = (char*)d_ws;
  auto alloc = [&](size_t bytes) { char* p = w; w += (bytes + 255) & ~(size_t)255; return p; };
  __bf16* xqb = (__bf16*)alloc(NX * 2);
  __bf16* xkb = (__bf16*)alloc(NX * 2);
  __bf16* xvb = (__bf16*)alloc(NX * 2);
  __bf16* Wqb = (__bf16*)alloc(NW * 2);
  __bf16* Wkb = (__bf16*)alloc(NW * 2);
  __bf16* Wvb = (__bf16*)alloc(NW * 2);
  __bf16* Wob = (__bf16*)alloc(NW * 2);
  __bf16* qb  = (__bf16*)alloc(NX * 2);
  __bf16* kb  = (__bf16*)alloc(NX * 2);
  __bf16* vb  = (__bf16*)alloc(NX * 2);
  __bf16* vT  = (__bf16*)alloc(NX * 2);                  // [B*H, D, S]
  __bf16* ctx = (__bf16*)alloc(NX * 2);                  // [B,S,F]
  float*  ypr = (float*)alloc(NX * 4);                   // out-proj f32

  dim3 blk(256);

  // 1) f32 -> bf16 conversions
  {
    int gx = (int)(NX / 4 / 256);
    cvt_f32_bf16<<<gx, blk, 0, stream>>>(query, xqb, (int)(NX / 4));
    cvt_f32_bf16<<<gx, blk, 0, stream>>>(key,   xkb, (int)(NX / 4));
    cvt_f32_bf16<<<gx, blk, 0, stream>>>(value, xvb, (int)(NX / 4));
    int gw = (int)(NW / 4 / 256);
    cvt_f32_bf16<<<gw, blk, 0, stream>>>(Wq, Wqb, (int)(NW / 4));
    cvt_f32_bf16<<<gw, blk, 0, stream>>>(Wk, Wkb, (int)(NW / 4));
    cvt_f32_bf16<<<gw, blk, 0, stream>>>(Wv, Wvb, (int)(NW / 4));
    cvt_f32_bf16<<<gw, blk, 0, stream>>>(Wo, Wob, (int)(NW / 4));
  }

  // 2) Q/K/V projections: [8192,1024] x [1024,1024]^T + bias -> bf16
  {
    dim3 g(Fsz / 128, (Bsz * Ssz) / 128, 1);
    gemm_nt_wmma<128,128,2,4,false,false,true><<<g, blk, 0, stream>>>(
        xqb, Wqb, bq, qb, Fsz, Fsz, Fsz, Fsz, 0,0,0,0,0,0, 1, 1.0f);
    gemm_nt_wmma<128,128,2,4,false,false,true><<<g, blk, 0, stream>>>(
        xkb, Wkb, bk, kb, Fsz, Fsz, Fsz, Fsz, 0,0,0,0,0,0, 1, 1.0f);
    gemm_nt_wmma<128,128,2,4,false,false,true><<<g, blk, 0, stream>>>(
        xvb, Wvb, bv, vb, Fsz, Fsz, Fsz, Fsz, 0,0,0,0,0,0, 1, 1.0f);
  }

  // 3) vT[bh][d][s] for the P@V NT-GEMM
  {
    dim3 g(Ssz / 32, Dsz / 32, Bsz * Hsz);
    transpose_v<<<g, blk, 0, stream>>>(vb, vT);
  }

  // 4) scores = 0.5 * Q @ K^T -> attn region (f32), 128 heads batched
  {
    dim3 g(Ssz / 128, Ssz / 128, Bsz * Hsz);
    gemm_nt_wmma<128,128,2,4,false,true,false><<<g, blk, 0, stream>>>(
        qb, kb, nullptr, attn, Fsz, Fsz, Ssz, Dsz,
        (long long)Ssz * Fsz, Dsz,                          // A: head offset into q
        (long long)Ssz * Fsz, Dsz,                          // B: head offset into k
        (long long)Hsz * Ssz * Ssz, (long long)Ssz * Ssz,   // C: z*S*S
        Hsz, SCALE);
  }

  // 5) row softmax, in place
  {
    long long rows = (long long)Bsz * Hsz * Ssz;
    softmax_rows<<<(int)(rows / 8), blk, 0, stream>>>(attn, rows);
  }

  // 6) context = P @ V : A = attn f32 (converted while staging), B = vT bf16
  {
    dim3 g(1, Ssz / 128, Bsz * Hsz);
    gemm_nt_wmma<128,64,4,2,true,false,false><<<g, blk, 0, stream>>>(
        attn, vT, nullptr, ctx, Ssz, Ssz, Fsz, Ssz,
        (long long)Hsz * Ssz * Ssz, (long long)Ssz * Ssz,   // A: z*S*S
        (long long)Hsz * Dsz * Ssz, (long long)Dsz * Ssz,   // B: z*D*S
        (long long)Ssz * Fsz, Dsz,                          // C: [B,S,F] head cols
        Hsz, 1.0f);
  }

  // 7) output projection: ctx @ Wo^T + bo -> f32
  {
    dim3 g(Fsz / 128, (Bsz * Ssz) / 128, 1);
    gemm_nt_wmma<128,128,2,4,false,true,true><<<g, blk, 0, stream>>>(
        ctx, Wob, bo, ypr, Fsz, Fsz, Fsz, Fsz, 0,0,0,0,0,0, 1, 1.0f);
  }

  // 8) out = LayerNorm(query + ypr) * gamma + beta
  {
    long long rows = (long long)Bsz * Ssz;
    residual_layernorm<<<(int)(rows / 8), blk, 0, stream>>>(query, ypr, gamma, beta, out, rows);
  }
}
